// AFTFull_16870631539008
// MI455X (gfx1250) — matrix operand order repaired and checked by reading, not compiled, not run on hardware
//
#include <hip/hip_runtime.h>
#include <hip/hip_bf16.h>
#include <stddef.h>

// ---------------------------------------------------------------------------
// AFT-Full on MI455X (gfx1250, wave32).
// exp_pos_bias == 1 (max over singleton dim) -> num/den are j-sums independent
// of i. Heavy work = 4 GEMMs (8192x1024x1024), run as bf16 WMMA with f32
// accumulation. Tile staging uses the Tensor Data Mover (tensor_load_to_lds,
// TENSORcnt, double-buffered LDS) instead of VALU/VMEM staging.
// ---------------------------------------------------------------------------

typedef __attribute__((ext_vector_type(16))) __bf16        v16bf;
typedef __attribute__((ext_vector_type(8)))  float         v8f;
typedef __attribute__((ext_vector_type(4)))  unsigned int  v4u;
typedef __attribute__((ext_vector_type(8)))  int           v8i;
typedef __attribute__((ext_vector_type(4)))  int           v4i;

__device__ __forceinline__ unsigned short f32_to_bf16_rne(float f) {
    union { float f; unsigned int u; } v;
    v.f = f;
    unsigned int u = v.u;
    u += 0x7FFFu + ((u >> 16) & 1u);   // round-to-nearest-even
    return (unsigned short)(u >> 16);
}

__device__ __forceinline__ __bf16 us_as_bf16(unsigned short u) {
    union { unsigned short u; __bf16 b; } cv;
    cv.u = u;
    return cv.b;
}

__device__ __forceinline__ unsigned lds_off(const void* p) {
    // Generic LDS pointers carry the wave-relative LDS byte offset in the
    // low 32 bits (ISA 10.2 aperture mapping).
    return (unsigned)(size_t)p;
}

// ---------------------------------------------------------------------------
// TDM: DMA one 128-row x 32-col bf16 tile (row-major, row stride Kd elements)
// from global memory into LDS. LDS padding: +1 dword after every 16 dwords
// (one 64B row) -> destination row stride = 34 ushorts (bank-conflict free).
// Descriptor per ISA 8.3/8.4 (D# group0 + group1; groups 2/3 zero -> 2D).
// ---------------------------------------------------------------------------
__device__ __forceinline__ void tdm_load_tile_bf16(const unsigned short* gsrc,
                                                   unsigned lds_byte,
                                                   int Kd, int tensor_rows) {
    const unsigned long long ga = (unsigned long long)(size_t)gsrc;
    v4u g0;
    g0.x = 1u;                                            // count=1 (valid), user mode
    g0.y = lds_byte;                                      // lds_addr
    g0.z = (unsigned)ga;                                  // global_addr[31:0]
    g0.w = ((unsigned)(ga >> 32) & 0x01FFFFFFu)           // global_addr[56:32]
           | (2u << 30);                                  // type = 2 ("image")
    v8i g1;
    g1[0] = (int)((1u << 16)                              // data_size = 1 (2 bytes)
                  | (1u << 20)                            // pad_enable
                  | (3u << 22));                          // pad_interval = 3 (16 dwords)
                                                          // pad_amount = 0 (+1 dword)
    g1[1] = (int)(((unsigned)Kd & 0xFFFFu) << 16);        // tensor_dim0[15:0]
    g1[2] = (int)((((unsigned)Kd >> 16) & 0xFFFFu)        // tensor_dim0[31:16]
                  | (((unsigned)tensor_rows & 0xFFFFu) << 16));   // tensor_dim1[15:0]
    g1[3] = (int)((((unsigned)tensor_rows >> 16) & 0xFFFFu)       // tensor_dim1[31:16]
                  | (32u << 16));                         // tile_dim0 = 32
    g1[4] = 128;                                          // tile_dim1 = 128, tile_dim2 = 0
    g1[5] = Kd;                                           // tensor_dim0_stride[31:0]
    g1[6] = 0;                                            // stride hi / dim1_stride lo
    g1[7] = 0;
    const v4i z4 = {0, 0, 0, 0};
#if __clang_major__ >= 23
    const v8i z8 = {0, 0, 0, 0, 0, 0, 0, 0};
    __builtin_amdgcn_tensor_load_to_lds(g0, g1, z4, z4, z8, 0);
#else
    __builtin_amdgcn_tensor_load_to_lds(g0, g1, z4, z4, 0);
#endif
}

// ---------------------------------------------------------------------------
// Y[r][n] = sum_k X[r][k] * W[n][k]   (X: (R,Kd) bf16, W: (N,Kd) bf16, Y f32)
// Block tile 128x128x32, 256 threads = 8 waves, each wave owns 64x32 = 4x2
// WMMA 16x16 tiles. Double-buffered LDS fed by the TDM.
// ---------------------------------------------------------------------------
#define BM 128
#define BN 128
#define BK 32
#define LDSK 34   // 32 bf16 + 1-dword TDM pad -> 17-dword row stride

__global__ __launch_bounds__(256)
void gemm_bf16_tdm(const unsigned short* __restrict__ X,
                   const unsigned short* __restrict__ W,
                   float* __restrict__ Y,
                   int R, int N, int Kd) {
    __shared__ unsigned short As[2][BM][LDSK];
    __shared__ unsigned short Bs[2][BN][LDSK];

    const int t    = threadIdx.x;
    const int lane = t & 31;
    const int wave = t >> 5;          // 0..7
    const int wm   = wave & 1;        // 2 waves along M (64 rows each)
    const int wn   = wave >> 1;       // 4 waves along N (32 cols each)

    const int m0 = blockIdx.y * BM;
    const int n0 = blockIdx.x * BN;
    const int nsteps = Kd / BK;

    // Preload buffer 0 via TDM (wave 0 issues; EXEC-independent DMA).
    if (t < 32) {
        tdm_load_tile_bf16(X + (size_t)m0 * Kd, lds_off(&As[0][0][0]), Kd, R);
        tdm_load_tile_bf16(W + (size_t)n0 * Kd, lds_off(&Bs[0][0][0]), Kd, N);
    }

    v8f acc[4][2];
    const v8f vzero = {0.f, 0.f, 0.f, 0.f, 0.f, 0.f, 0.f, 0.f};
#pragma unroll
    for (int i = 0; i < 4; ++i)
#pragma unroll
        for (int j = 0; j < 2; ++j) acc[i][j] = vzero;

    for (int s = 0; s < nsteps; ++s) {
        const int cur = s & 1;
        const bool has_next = (s + 1) < nsteps;

        if (t < 32) {
            if (has_next) {
                const int kb2 = (s + 1) * BK;
                tdm_load_tile_bf16(X + (size_t)m0 * Kd + kb2,
                                   lds_off(&As[cur ^ 1][0][0]), Kd, R);
                tdm_load_tile_bf16(W + (size_t)n0 * Kd + kb2,
                                   lds_off(&Bs[cur ^ 1][0][0]), Kd, N);
                __builtin_amdgcn_s_wait_tensorcnt(2);   // current pair retired
            } else {
                __builtin_amdgcn_s_wait_tensorcnt(0);   // drain
            }
        }
        __syncthreads();   // LDS buffer `cur` visible to all waves

        // ---- fragments per ISA 7.12.2 layouts ----
        const int mrl   = lane & 15;
        const int hi    = lane >> 4;          // lane half
        const int kbase = hi ? 8 : 0;         // A-frag K interleave base
        const int kh    = hi << 4;            // unused; B-frag also indexes via kbase

        v16bf afr[4];
#pragma unroll
        for (int i = 0; i < 4; ++i) {
            const int mr = wm * 64 + i * 16 + mrl;
#pragma unroll
            for (int e = 0; e < 16; ++e) {
                const int kk = kbase + ((e < 8) ? e : (e + 8));
                afr[i][e] = us_as_bf16(As[cur][mr][kk]);
            }
        }
        v16bf bfr[2];
#pragma unroll
        for (int j = 0; j < 2; ++j) {
            const int nc = wn * 32 + j * 16 + mrl;
#pragma unroll
            for (int e = 0; e < 16; ++e)
                bfr[j][e] = us_as_bf16(Bs[cur][nc][kbase + ((e < 8) ? e : (e + 8))]);
        }

        // ---- 8 WMMAs: D = A*B + C (f32 accumulate) ----
#pragma unroll
        for (int i = 0; i < 4; ++i)
#pragma unroll
            for (int j = 0; j < 2; ++j)
                acc[i][j] = __builtin_amdgcn_wmma_f32_16x16x32_bf16(
                    false, afr[i], false, bfr[j],
                    (short)0, acc[i][j], false, false);

        __syncthreads();   // everyone done with buffer `cur` before next DMA overwrites it
    }

    // ---- write back: C/D 16x16 f32 layout (ISA 7.12.2) ----
    const int mrl = lane & 15;
    const int mhi = (lane >> 4) * 8;
#pragma unroll
    for (int i = 0; i < 4; ++i) {
#pragma unroll
        for (int j = 0; j < 2; ++j) {
            const int ncol = n0 + wn * 32 + j * 16 + mrl;
#pragma unroll
            for (int v = 0; v < 8; ++v) {
                const int mrow = m0 + wm * 64 + i * 16 + v + mhi;
                Y[(size_t)mrow * N + ncol] = acc[i][j][v];
            }
        }
    }
}

// ---------------------------------------------------------------------------
// One-shot f32 -> bf16 conversion (4 elements per thread).
// ---------------------------------------------------------------------------
__global__ __launch_bounds__(256)
void cvt_f32_bf16(const float* __restrict__ in, unsigned short* __restrict__ out) {
    const size_t i = ((size_t)blockIdx.x * blockDim.x + threadIdx.x) * 4;
    const float4 f = *(const float4*)(in + i);
    const unsigned int lo = (unsigned)f32_to_bf16_rne(f.x) | ((unsigned)f32_to_bf16_rne(f.y) << 16);
    const unsigned int hi = (unsigned)f32_to_bf16_rne(f.z) | ((unsigned)f32_to_bf16_rne(f.w) << 16);
    *(uint2*)(out + i) = make_uint2(lo, hi);
}

// ---------------------------------------------------------------------------
// ratio[b][d] = sum_j e(b,j,d)*V[b,j,d] / sum_j e(b,j,d),
// e = exp(K[b,j,d] - max_b' K[b',j,d])
// ---------------------------------------------------------------------------
__global__ __launch_bounds__(256)
void aft_reduce(const float* __restrict__ Kp, const float* __restrict__ Vp,
                float* __restrict__ ratio, int T, int D) {
    const int idx = blockIdx.x * blockDim.x + threadIdx.x;   // 0 .. 4*D-1
    const int b = idx / D;
    const int d = idx - b * D;
    const size_t TD = (size_t)T * D;
    float s0 = 0.f, s1 = 0.f;
    for (int j = 0; j < T; ++j) {
        const size_t off = (size_t)j * D + d;
        const float k0 = Kp[off];
        const float k1 = Kp[TD + off];
        const float k2 = Kp[2 * TD + off];
        const float k3 = Kp[3 * TD + off];
        const float m  = fmaxf(fmaxf(k0, k1), fmaxf(k2, k3));
        const float e  = __expf(Kp[(size_t)b * TD + off] - m);
        s0 += e;
        s1 += e * Vp[(size_t)b * TD + off];
    }
    ratio[idx] = s1 / s0;
}

// Yt = sigmoid(Q) * ratio[b,d], emitted directly as bf16 for the output GEMM.
__global__ __launch_bounds__(256)
void aft_yt(const float* __restrict__ Q, unsigned short* __restrict__ Yt,
            const float* __restrict__ ratio, int D, int TD_per_batch) {
    const size_t i = (size_t)blockIdx.x * blockDim.x + threadIdx.x;
    const float x = Q[i];
    const float sig = 1.0f / (1.0f + __expf(-x));
    const int d = (int)(i % D);
    const int b = (int)(i / (size_t)TD_per_batch);
    Yt[i] = f32_to_bf16_rne(sig * ratio[b * D + d]);
}

// ---------------------------------------------------------------------------
extern "C" void kernel_launch(void* const* d_in, const int* in_sizes, int n_in,
                              void* d_out, int out_size, void* d_ws, size_t ws_size,
                              hipStream_t stream) {
    (void)in_sizes; (void)n_in; (void)out_size; (void)ws_size;

    const int B = 4, T = 2048, DM = 1024, DH = 1024;
    const int R = B * T;                       // 8192 rows
    const size_t RD = (size_t)R * DH;          // 8.4M elements
    const size_t WD = (size_t)DH * DM;         // 1M elements

    const float* q  = (const float*)d_in[0];
    const float* k  = (const float*)d_in[1];
    const float* v  = (const float*)d_in[2];
    const float* Wq = (const float*)d_in[3];
    const float* Wk = (const float*)d_in[4];
    const float* Wv = (const float*)d_in[5];
    const float* Wo = (const float*)d_in[6];
    // d_in[7] = W_bias: mathematically dead (exp(bias - max over singleton) == 1)

    char* ws = (char*)d_ws;
    float* Qf    = (float*)ws;                 ws += RD * sizeof(float);   // 32 MB
    float* Kf    = (float*)ws;                 ws += RD * sizeof(float);   // 32 MB
    float* Vf    = (float*)ws;                 ws += RD * sizeof(float);   // 32 MB
    float* ratio = (float*)ws;                 ws += (size_t)B * DH * sizeof(float);
    unsigned short* qb  = (unsigned short*)ws; ws += RD * 2;               // 16 MB
    unsigned short* kb  = (unsigned short*)ws; ws += RD * 2;
    unsigned short* vb  = (unsigned short*)ws; ws += RD * 2;
    unsigned short* wqb = (unsigned short*)ws; ws += WD * 2;               // 2 MB
    unsigned short* wkb = (unsigned short*)ws; ws += WD * 2;
    unsigned short* wvb = (unsigned short*)ws; ws += WD * 2;
    unsigned short* wob = (unsigned short*)ws; ws += WD * 2;
    unsigned short* ytb = (unsigned short*)ws; ws += RD * 2;               // 16 MB

    // 1) one-shot bf16 conversions
    const unsigned gin = (unsigned)(RD / (4 * 256));   // 8192 blocks
    const unsigned gwt = (unsigned)(WD / (4 * 256));   // 1024 blocks
    cvt_f32_bf16<<<gin, 256, 0, stream>>>(q,  qb);
    cvt_f32_bf16<<<gin, 256, 0, stream>>>(k,  kb);
    cvt_f32_bf16<<<gin, 256, 0, stream>>>(v,  vb);
    cvt_f32_bf16<<<gwt, 256, 0, stream>>>(Wq, wqb);
    cvt_f32_bf16<<<gwt, 256, 0, stream>>>(Wk, wkb);
    cvt_f32_bf16<<<gwt, 256, 0, stream>>>(Wv, wvb);
    cvt_f32_bf16<<<gwt, 256, 0, stream>>>(Wo, wob);

    // 2) projections (bf16 WMMA, TDM-staged)
    dim3 gg(DH / BN, R / BM), bb(256);
    gemm_bf16_tdm<<<gg, bb, 0, stream>>>(qb, wqb, Qf, R, DH, DM);
    gemm_bf16_tdm<<<gg, bb, 0, stream>>>(kb, wkb, Kf, R, DH, DM);
    gemm_bf16_tdm<<<gg, bb, 0, stream>>>(vb, wvb, Vf, R, DH, DM);

    // 3) degenerate AFT attention: per-(b,d) j-reductions
    aft_reduce<<<(B * DH) / 256, 256, 0, stream>>>(Kf, Vf, ratio, T, DH);

    // 4) Yt = sigmoid(Q) * ratio  (bf16 out)
    aft_yt<<<(unsigned)(RD / 256), 256, 0, stream>>>(Qf, ytb, ratio, DH, T * DH);

    // 5) output projection -> d_out (f32)
    dim3 go(DM / BN, R / BM);
    gemm_bf16_tdm<<<go, bb, 0, stream>>>(ytb, wob, (float*)d_out, R, DM, DH);
}
